// paren_mLSTM_74534862455047
// MI455X (gfx1250) — matrix operand order repaired and checked by reading, not compiled, not run on hardware
//
#include <hip/hip_runtime.h>

// Problem dims (fixed by the reference)
#define BB   128
#define SS   1024
#define EE   512
#define HH   512
#define KKC  2       // number of expert cells
#define G4H  2048    // 4*H

typedef __attribute__((ext_vector_type(16))) __bf16 v16bf;
typedef __attribute__((ext_vector_type(8)))  __bf16 v8bf;
typedef __attribute__((ext_vector_type(8)))  float  v8f;

__device__ __forceinline__ float sigmf(float x) { return 1.0f / (1.0f + __expf(-x)); }

// A fragment (16x32 bf16, M x K): lane holds row M = lane&15.
// VGPR layout (ISA 7.12.2): lanes 0-15 get K = [kb, kb+8) then [kb+16, kb+24) with kb = (lane>>4)*8.
__device__ __forceinline__ v16bf load_a_frag(const __bf16* row_k0, int kbA) {
  v8bf lo = *(const v8bf*)(row_k0 + kbA);
  v8bf hi = *(const v8bf*)(row_k0 + kbA + 16);
  v16bf a;
#pragma unroll
  for (int i = 0; i < 8; ++i) { a[i] = lo[i]; a[i + 8] = hi[i]; }
  return a;
}

// B fragment (32x16 bf16, K x N): lane holds column N = lane&15;
// lanes 0-15 hold K=[0,8)+[16,24), lanes 16-31 hold K=[8,16)+[24,32) (same order as A).
__device__ __forceinline__ v16bf load_b_frag(const __bf16* wrow_k0) {
  v8bf lo = *(const v8bf*)(wrow_k0);
  v8bf hi = *(const v8bf*)(wrow_k0 + 16);
  v16bf b;
#pragma unroll
  for (int i = 0; i < 8; ++i) { b[i] = lo[i]; b[i + 8] = hi[i]; }
  return b;
}

__global__ void f32_to_bf16_kernel(const float* __restrict__ src, __bf16* __restrict__ dst, int n) {
  int i = (blockIdx.x * blockDim.x + threadIdx.x) * 4;
  if (i + 3 < n) {
    float4 v = *(const float4*)(src + i);
    dst[i + 0] = (__bf16)v.x;
    dst[i + 1] = (__bf16)v.y;
    dst[i + 2] = (__bf16)v.z;
    dst[i + 3] = (__bf16)v.w;
  } else {
    for (; i < n; ++i) dst[i] = (__bf16)src[i];
  }
}

__global__ void bias_add_kernel(const float* __restrict__ a, const float* __restrict__ b,
                                float* __restrict__ o, int n) {
  int i = blockIdx.x * blockDim.x + threadIdx.x;
  if (i < n) o[i] = a[i] + b[i];
}

__global__ void init_state_kernel(__bf16* __restrict__ h, float* __restrict__ c, int n) {
  int i = blockIdx.x * blockDim.x + threadIdx.x;
  if (i < n) {
    c[i] = 0.0f;
    h[i] = (__bf16)0.0f;
    h[i + n] = (__bf16)0.0f;   // both ping-pong h buffers
  }
}

// One timestep: grid = (H/16 j-tiles, B/16 batch-tiles), 256 threads (8 waves).
// Wave w -> (cell = w&1, gate = w>>1); computes a 16x16 f32 gate tile via 32 bf16 WMMAs.
__global__ __launch_bounds__(256) void lstm_step_kernel(
    const __bf16* __restrict__ X,     // [B,S,E] bf16
    const __bf16* __restrict__ Wih,   // [K,4H,E] bf16
    const __bf16* __restrict__ Whh,   // [K,4H,H] bf16
    const float*  __restrict__ bias,  // [K,4H]
    const int*    __restrict__ inp,   // [B,S]
    const __bf16* __restrict__ h_in,  // [B,H] bf16 (t-1)
    __bf16* __restrict__ h_out,       // [B,H] bf16 (t)
    float* __restrict__ cbuf,         // [B,H] f32 state
    float* __restrict__ out,          // [B,S,H] f32
    int t)
{
  __shared__ float lds_g[KKC][4][16][17];   // +1 pad: no LDS bank conflicts

  const int j0 = blockIdx.x * 16;   // position within H
  const int b0 = blockIdx.y * 16;   // position within B
  const int tid  = threadIdx.x;
  const int wave = tid >> 5;
  const int lane = tid & 31;

  // -------- WMMA phase --------
  {
    const int cell = wave & 1;
    const int gate = wave >> 1;           // 0..3 -> i, f, g, o
    const int nloc = lane & 15;           // A row (M) and B column (N)
    const int half = lane >> 4;
    const int kbA  = half * 8;
    const int kbB  = half * 8;
    const int grow = gate * HH + j0 + nloc;   // W row within 4H

    const __bf16* wih_row = Wih + ((size_t)cell * G4H + grow) * EE + kbB;
    const __bf16* whh_row = Whh + ((size_t)cell * G4H + grow) * HH + kbB;
    const __bf16* x_row   = X + ((size_t)(b0 + nloc) * SS + t) * EE;
    const __bf16* h_row   = h_in + (size_t)(b0 + nloc) * HH;

    v8f acc = {};
    // x contribution: reduce over E
#pragma unroll 4
    for (int kk = 0; kk < EE; kk += 32) {
      v16bf a = load_a_frag(x_row + kk, kbA);
      v16bf b = load_b_frag(wih_row + kk);
      acc = __builtin_amdgcn_wmma_f32_16x16x32_bf16(false, a, false, b, (short)0, acc, false, false);
    }
    // recurrent contribution: reduce over H
#pragma unroll 4
    for (int kk = 0; kk < HH; kk += 32) {
      v16bf a = load_a_frag(h_row + kk, kbA);
      v16bf b = load_b_frag(whh_row + kk);
      acc = __builtin_amdgcn_wmma_f32_16x16x32_bf16(false, a, false, b, (short)0, acc, false, false);
    }

    const float bv = bias[(size_t)cell * G4H + grow];
    // D layout: VGPR r -> row M = r + half*8, col N = lane&15
#pragma unroll
    for (int r = 0; r < 8; ++r)
      lds_g[cell][gate][r + half * 8][nloc] = acc[r] + bv;
  }

  __syncthreads();

  // -------- routed selection + LSTM pointwise --------
  {
    const int m = tid >> 4;
    const int n = tid & 15;
    const int b = b0 + m;
    const int j = j0 + n;
    const int ksel = inp[(size_t)b * SS + t] & 1;   // vocab id % 2 (ids are non-negative)

    const float gi = lds_g[ksel][0][m][n];
    const float gf = lds_g[ksel][1][m][n];
    const float gg = lds_g[ksel][2][m][n];
    const float go = lds_g[ksel][3][m][n];

    const float c_old = cbuf[(size_t)b * HH + j];
    const float c_new = sigmf(gf) * c_old + sigmf(gi) * tanhf(gg);
    const float h_new = sigmf(go) * tanhf(c_new);

    cbuf[(size_t)b * HH + j] = c_new;
    out[((size_t)b * SS + t) * HH + j] = h_new;
    h_out[(size_t)b * HH + j] = (__bf16)h_new;
  }
}

__global__ void finalize_kernel(const float* __restrict__ out, const float* __restrict__ cbuf,
                                float* __restrict__ hlast, float* __restrict__ clast) {
  int i = blockIdx.x * blockDim.x + threadIdx.x;   // n = B*H
  if (i < BB * HH) {
    int b = i / HH, j = i % HH;
    hlast[i] = out[((size_t)b * SS + (SS - 1)) * HH + j];
    clast[i] = cbuf[i];
  }
}

extern "C" void kernel_launch(void* const* d_in, const int* in_sizes, int n_in,
                              void* d_out, int out_size, void* d_ws, size_t ws_size,
                              hipStream_t stream)
{
  const int*   inp  = (const int*)  d_in[0];   // [B,S] int32
  const float* xemb = (const float*)d_in[1];   // [B,S,E] f32
  const float* W_ih = (const float*)d_in[2];   // [K,4H,E] f32
  const float* W_hh = (const float*)d_in[3];   // [K,4H,H] f32
  const float* b_ih = (const float*)d_in[4];   // [K,4H] f32
  const float* b_hh = (const float*)d_in[5];   // [K,4H] f32
  float* out = (float*)d_out;                  // [B,S,H] ++ [B,H] ++ [B,H] f32

  // Workspace carve-up (~143 MB, all regions power-of-two aligned)
  char* ws = (char*)d_ws;
  size_t o = 0;
  __bf16* Xb    = (__bf16*)(ws + o); o += (size_t)BB * SS * EE * 2;
  __bf16* Wihb  = (__bf16*)(ws + o); o += (size_t)KKC * G4H * EE * 2;
  __bf16* Whhb  = (__bf16*)(ws + o); o += (size_t)KKC * G4H * HH * 2;
  float*  biasb = (float*) (ws + o); o += (size_t)KKC * G4H * 4;
  __bf16* hbuf  = (__bf16*)(ws + o); o += (size_t)2 * BB * HH * 2;   // ping-pong h
  float*  cbuf  = (float*) (ws + o); o += (size_t)BB * HH * 4;

  // Phase 1: precision conversion + bias combine + state init (all parallel, all on stream)
  {
    int n = BB * SS * EE;
    f32_to_bf16_kernel<<<(n / 4 + 255) / 256, 256, 0, stream>>>(xemb, Xb, n);
    n = KKC * G4H * EE;
    f32_to_bf16_kernel<<<(n / 4 + 255) / 256, 256, 0, stream>>>(W_ih, Wihb, n);
    n = KKC * G4H * HH;
    f32_to_bf16_kernel<<<(n / 4 + 255) / 256, 256, 0, stream>>>(W_hh, Whhb, n);
    n = KKC * G4H;
    bias_add_kernel<<<(n + 255) / 256, 256, 0, stream>>>(b_ih, b_hh, biasb, n);
    n = BB * HH;
    init_state_kernel<<<(n + 255) / 256, 256, 0, stream>>>(hbuf, cbuf, n);
  }

  // Phase 2: sequential scan, one kernel per timestep (graph replay amortizes launches).
  // h is ping-ponged by t parity so no block reads h(t) while another writes it.
  dim3 grid(HH / 16, BB / 16);
  for (int t = 0; t < SS; ++t) {
    const __bf16* hin = hbuf + (size_t)(t & 1) * BB * HH;
    __bf16* hout      = hbuf + (size_t)((t + 1) & 1) * BB * HH;
    lstm_step_kernel<<<grid, 256, 0, stream>>>(Xb, Wihb, Whhb, biasb, inp,
                                               hin, hout, cbuf, out, t);
  }

  // Phase 3: emit h_last, c_last into the output tail
  finalize_kernel<<<(BB * HH + 255) / 256, 256, 0, stream>>>(
      out, cbuf, out + (size_t)BB * SS * HH, out + (size_t)BB * SS * HH + (size_t)BB * HH);

  (void)in_sizes; (void)n_in; (void)out_size; (void)ws_size;
}
